// ESRNN_model_63634235457548
// MI455X (gfx1250) — matrix operand order repaired and checked by reading, not compiled, not run on hardware
//
#include <hip/hip_runtime.h>
#include <hip/hip_bf16.h>

// ES-RNN for MI455X (gfx1250, wave32).
// All matmuls use v_wmma_f32_16x16x32_bf16 (fp32 accumulate).
// LSTM layers are persistent kernels: one workgroup owns 16 batch rows,
// keeps c/h in LDS across timesteps, streams pre-packed bf16 weight
// fragments from L2 (<=1MB/layer << 192MB L2) with double-buffered,
// immediate-offset b128 loads (KT is a template constant).

typedef __attribute__((ext_vector_type(16))) __bf16 bf16x16;
typedef __attribute__((ext_vector_type(8)))  float  f32x8;
typedef __attribute__((ext_vector_type(4)))  float  f32x4;

#define TLEN 673   // effective sequence length
#define NSER 256   // batch (series)
#define HID  256

union Frag { bf16x16 v; f32x4 q[2]; };

__device__ __forceinline__ float sigf(float x) { return 1.0f / (1.0f + __expf(-x)); }

// ---------------------------------------------------------------------------
// 1) Holt-Winters smoothing: one thread per series, sequential scan over T.
// ---------------------------------------------------------------------------
__global__ void hw_kernel(const float* __restrict__ inputs,
                          const int*   __restrict__ idxs,
                          const float* __restrict__ level_coef,
                          const float* __restrict__ season_coef,
                          const float* __restrict__ seas_param,
                          float* __restrict__ levels,
                          float* __restrict__ seas)
{
    __shared__ float ring[256 * 25];
    const int b  = threadIdx.x;
    const int id = idxs[b];
    const float a = sigf(level_coef[id]);
    const float g = sigf(season_coef[id]);
    float* rb = &ring[b * 25];

    const float si0 = __expf(seas_param[id * 24 + 0]);
    for (int k = 0; k < 24; ++k) {
        float sv = __expf(seas_param[id * 24 + k]);
        seas[b * 744 + k] = sv;
        if (k >= 1) rb[k - 1] = sv;       // buf0 = seas_init[1..24]
    }
    seas[b * 744 + 24] = si0;             // seas_init[24] = seas_init[0]
    rb[23] = si0;

    const float seas1 = rb[0];            // constant throughout scan (per ref)
    float lev = inputs[b * 720 + 0] / si0;
    levels[b * 720 + 0] = lev;
    int head = 0;
    for (int t = 1; t < 720; ++t) {
        float x = inputs[b * 720 + t];
        lev = a * (x / seas1) + (1.0f - a) * lev;
        float ns = g * (x / lev) + (1.0f - g) * rb[head];
        rb[head] = ns;
        head = (head + 1) % 24;
        levels[b * 720 + t] = lev;
        seas[b * 744 + 24 + t] = ns;
    }
}

// ---------------------------------------------------------------------------
// 2) Window normalization: x (bf16, (673,256,24)) and out_list -> d_out
// ---------------------------------------------------------------------------
__global__ void window_kernel(const float* __restrict__ inputs,
                              const float* __restrict__ levels,
                              const float* __restrict__ seas,
                              __bf16* __restrict__ xout,
                              float*  __restrict__ actual)
{
    const int t = blockIdx.x;     // 0..672
    const int b = threadIdx.x;    // 0..255
    const float lev = levels[b * 720 + t + 23];
    const float* inp = inputs + b * 720;
    const float* se  = seas   + b * 744;
    const size_t base = ((size_t)t * NSER + b) * 24;
    for (int j = 0; j < 24; ++j) {
        xout[base + j]   = (__bf16)(inp[t + j] / se[t + j] / lev);
        actual[base + j] = inp[24 + t + j] / se[24 + t + j] / lev;
    }
}

// ---------------------------------------------------------------------------
// 3) Weight packing: W = [Wih | Whh] as B-matrix (Ktot x Nout), bf16,
//    fragment-major: [ntile][ktile][lane][16], lane: N = lane%16,
//    K-run = ktile*32 + (lane/16)*16 .. +15  (contiguous 32B per lane)
// ---------------------------------------------------------------------------
__global__ void pack_kernel(const float* __restrict__ Wih, int C,
                            const float* __restrict__ Whh, int Kxp,
                            int KT, int ntiles, int NoutReal,
                            __bf16* __restrict__ dst)
{
    const int gid = blockIdx.x * blockDim.x + threadIdx.x;
    const int total = ntiles * KT * 32;
    if (gid >= total) return;
    const int lane = gid & 31;
    const int blk  = gid >> 5;
    const int nt = blk / KT, kt = blk % KT;
    const int n  = nt * 16 + (lane & 15);
    const int kb = kt * 32 + (lane >> 4) * 16;
    __bf16* d = dst + (size_t)blk * 512 + lane * 16;
    for (int j = 0; j < 16; ++j) {
        int k = kb + j;
        float v = 0.0f;
        if (n < NoutReal) {
            if (k < Kxp) { if (k < C) v = Wih[(size_t)n * C + k]; }
            else         { v = Whh[(size_t)n * HID + (k - Kxp)]; }
        }
        d[j] = (__bf16)v;
    }
}

// ---------------------------------------------------------------------------
// 4) Persistent dilated LSTM layer (KXP template const -> KT const ->
//    all fragment loads use immediate offsets; double-buffered B frags).
//    Workgroup = 16 batch rows, 8 waves; wave w owns hidden cols
//    w*32..+31 across all 4 gates -> gate math fully in registers.
// ---------------------------------------------------------------------------
template <int KXP>
__global__ void __launch_bounds__(256) lstm_kernel(
    const __bf16* __restrict__ in,
    const __bf16* __restrict__ Wp, const float* __restrict__ bias,
    __bf16* __restrict__ out, int dil)
{
    constexpr int C  = (KXP == 32) ? 24 : 256;   // real input width
    constexpr int KT = (KXP + HID) >> 5;         // k-tiles (9 or 16)

    __shared__ __align__(16) __bf16 Abuf[16 * 520];  // [row][KXP x | 256 h]
    __shared__ __align__(16) float  cbuf[16 * 260];  // cell state fp32

    const int tid  = threadIdx.x;
    const int lane = tid & 31;
    const int w    = tid >> 5;
    const int r0   = blockIdx.x * 16;
    const int jg   = r0 >> 8;         // dilation phase (B=256, 16 | 256)
    const int b0   = r0 & 255;

    for (int i = tid; i < 16 * 520; i += 256) Abuf[i] = (__bf16)0.0f;
    for (int i = tid; i < 16 * 260; i += 256) cbuf[i] = 0.0f;
    __syncthreads();

    const int nsteps = (TLEN - jg + dil - 1) / dil;
    const int arow   = lane & 15;
    const int khalf  = (lane >> 4) * 16;
    const int ncol   = lane & 15;
    const int mbase  = (lane >> 4) * 8;

    // per-thread constants: bias values for the two owned column subtiles
    const int hc0 = w * 32 + ncol, hc1 = hc0 + 16;
    const float bi0 = bias[hc0],       bi1 = bias[hc1];
    const float bf0 = bias[256 + hc0], bf1 = bias[256 + hc1];
    const float bg0 = bias[512 + hc0], bg1 = bias[512 + hc1];
    const float bo0 = bias[768 + hc0], bo1 = bias[768 + hc1];

    // wave/lane-folded base pointers (all per-fragment offsets are immediates)
    const __bf16* wbase = Wp + (size_t)lane * 16 + (size_t)(w * 2) * KT * 512;
    const __bf16* abase = &Abuf[arow * 520 + khalf];

    f32x8 acc[8];

    for (int s = 0; s < nsteps; ++s) {
        const int t = s * dil + jg;     // always < TLEN
        // --- stage A: copy x_t into A-buffer (x part), vector b128 ---
        if (KXP == 256) {
            const int rr = tid >> 4, cc = (tid & 15) * 16;
            const f32x4* src = (const f32x4*)&in[((size_t)t * NSER + b0 + rr) * 256 + cc];
            f32x4 v0 = src[0], v1 = src[1];
            *(f32x4*)&Abuf[rr * 520 + cc]     = v0;
            *(f32x4*)&Abuf[rr * 520 + cc + 8] = v1;
        } else {
            if (tid < 48) {   // 16 rows x 24 bf16 = 48 x (8 bf16 chunks)
                const int rr = tid / 3, cc = (tid % 3) * 8;
                f32x4 v = *(const f32x4*)&in[((size_t)t * NSER + b0 + rr) * (size_t)C + cc];
                *(f32x4*)&Abuf[rr * 520 + cc] = v;
            }
        }
        __syncthreads();

        // --- stage B: z = [x,h] @ W^T : double-buffered fragment pipeline ---
        #pragma unroll
        for (int i = 0; i < 8; ++i) { f32x8 z = {}; acc[i] = z; }

        Frag a0, a1, bA[8], bB[8];

        auto loadB = [&](int kt, Frag* B) {
            const __bf16* wkt = wbase + kt * 512;
            #pragma unroll
            for (int i = 0; i < 8; ++i) {
                const int off = ((i >> 1) * 16 + (i & 1)) * (KT * 512);
                B[i].q[0] = *(const f32x4*)(wkt + off);
                B[i].q[1] = *(const f32x4*)(wkt + off + 8);
            }
        };
        auto loadA = [&](int kt, Frag& A) {
            const __bf16* ap = abase + kt * 32;
            A.q[0] = *(const f32x4*)(ap);
            A.q[1] = *(const f32x4*)(ap + 8);
        };
        auto wmma8 = [&](Frag& A, Frag* B) {
            #pragma unroll
            for (int i = 0; i < 8; ++i)
                acc[i] = __builtin_amdgcn_wmma_f32_16x16x32_bf16(
                    false, A.v, false, B[i].v, (short)0, acc[i], false, false);
        };

        loadB(0, bA);
        #pragma unroll
        for (int kt = 0; kt + 2 <= KT; kt += 2) {
            loadA(kt, a0);
            loadB(kt + 1, bB);
            wmma8(a0, bA);
            loadA(kt + 1, a1);
            if (kt + 2 < KT) loadB(kt + 2, bA);
            wmma8(a1, bB);
        }
        if (KT & 1) { loadA(KT - 1, a0); wmma8(a0, bA); }
        __syncthreads();

        // --- stage C: gates + c/h update (registers + LDS c) ---
        __bf16* outp = out + ((size_t)t * NSER + b0 + mbase) * HID;
        #pragma unroll
        for (int ci = 0; ci < 2; ++ci) {
            const int hcol = (ci == 0) ? hc0 : hc1;
            const float bi = (ci == 0) ? bi0 : bi1;
            const float bf = (ci == 0) ? bf0 : bf1;
            const float bg = (ci == 0) ? bg0 : bg1;
            const float bo = (ci == 0) ? bo0 : bo1;
            #pragma unroll
            for (int r = 0; r < 8; ++r) {
                const int row = mbase + r;
                float iv = sigf(acc[0 + ci][r] + bi);
                float fv = sigf(acc[2 + ci][r] + bf);
                float gv = tanhf(acc[4 + ci][r] + bg);
                float ov = sigf(acc[6 + ci][r] + bo);
                float c  = fv * cbuf[row * 260 + hcol] + iv * gv;
                cbuf[row * 260 + hcol] = c;
                __bf16 hb = (__bf16)(ov * tanhf(c));
                Abuf[row * 520 + KXP + hcol] = hb;
                outp[(size_t)r * HID + hcol] = hb;
            }
        }
        __syncthreads();
    }
}

// ---------------------------------------------------------------------------
// 5) hf = tanh((h3 + r) @ nlW^T + nlb)   M=172288, K=256, N=256
// ---------------------------------------------------------------------------
__global__ void __launch_bounds__(256) fuse_tanh_kernel(
    const __bf16* __restrict__ h3, const __bf16* __restrict__ r,
    const __bf16* __restrict__ Wp, const float* __restrict__ nlb,
    __bf16* __restrict__ hf)
{
    __shared__ __align__(16) __bf16 Abuf[16 * 264];
    const int tid = threadIdx.x, lane = tid & 31, w = tid >> 5;
    const size_t r0 = (size_t)blockIdx.x * 16;
    for (int i = tid; i < 16 * 256; i += 256) {
        int rr = i >> 8, k = i & 255;
        size_t gi = (r0 + rr) * 256 + k;
        Abuf[rr * 264 + k] = (__bf16)((float)h3[gi] + (float)r[gi]);
    }
    __syncthreads();
    const int arow = lane & 15, khalf = (lane >> 4) * 16, mbase = (lane >> 4) * 8;

    const __bf16* wbase = Wp + (size_t)lane * 16 + (size_t)(w * 2) * 8 * 512;
    const __bf16* abase = &Abuf[arow * 264 + khalf];
    f32x8 acc0 = {}, acc1 = {};
    #pragma unroll
    for (int kt = 0; kt < 8; ++kt) {
        Frag af, b0, b1;
        const __bf16* ap = abase + kt * 32;
        af.q[0] = *(const f32x4*)ap;  af.q[1] = *(const f32x4*)(ap + 8);
        const __bf16* wkt = wbase + kt * 512;
        b0.q[0] = *(const f32x4*)(wkt);               b0.q[1] = *(const f32x4*)(wkt + 8);
        b1.q[0] = *(const f32x4*)(wkt + 8 * 512);     b1.q[1] = *(const f32x4*)(wkt + 8 * 512 + 8);
        acc0 = __builtin_amdgcn_wmma_f32_16x16x32_bf16(false, af.v, false, b0.v, (short)0, acc0, false, false);
        acc1 = __builtin_amdgcn_wmma_f32_16x16x32_bf16(false, af.v, false, b1.v, (short)0, acc1, false, false);
    }
    #pragma unroll
    for (int ci = 0; ci < 2; ++ci) {
        const int col = (w * 2 + ci) * 16 + (lane & 15);
        const float bb = nlb[col];
        const f32x8& acc = ci ? acc1 : acc0;
        #pragma unroll
        for (int rr = 0; rr < 8; ++rr)
            hf[(r0 + mbase + rr) * 256 + col] = (__bf16)tanhf(acc[rr] + bb);
    }
}

// ---------------------------------------------------------------------------
// 6) pred = hf @ scoreW^T + scoreb   M=172288, K=256, N=24 (padded to 32)
// ---------------------------------------------------------------------------
__global__ void __launch_bounds__(256) score_kernel(
    const __bf16* __restrict__ hf, const __bf16* __restrict__ Wp,
    const float* __restrict__ scoreb, float* __restrict__ pred)
{
    const int tid = threadIdx.x, lane = tid & 31, w = tid >> 5;
    const size_t r0 = (size_t)blockIdx.x * 128 + (size_t)w * 16;
    const int arow = lane & 15, khalf = (lane >> 4) * 16, mbase = (lane >> 4) * 8;

    const __bf16* wbase = Wp + (size_t)lane * 16;
    const __bf16* abase = hf + (r0 + arow) * 256 + khalf;
    f32x8 acc0 = {}, acc1 = {};
    #pragma unroll
    for (int kt = 0; kt < 8; ++kt) {
        Frag af, b0, b1;
        const __bf16* ap = abase + kt * 32;
        af.q[0] = *(const f32x4*)ap;  af.q[1] = *(const f32x4*)(ap + 8);
        const __bf16* wkt = wbase + kt * 512;
        b0.q[0] = *(const f32x4*)(wkt);               b0.q[1] = *(const f32x4*)(wkt + 8);
        b1.q[0] = *(const f32x4*)(wkt + 8 * 512);     b1.q[1] = *(const f32x4*)(wkt + 8 * 512 + 8);
        acc0 = __builtin_amdgcn_wmma_f32_16x16x32_bf16(false, af.v, false, b0.v, (short)0, acc0, false, false);
        acc1 = __builtin_amdgcn_wmma_f32_16x16x32_bf16(false, af.v, false, b1.v, (short)0, acc1, false, false);
    }
    #pragma unroll
    for (int nt = 0; nt < 2; ++nt) {
        const int col = nt * 16 + (lane & 15);
        if (col < 24) {
            const float bb = scoreb[col];
            const f32x8& acc = nt ? acc1 : acc0;
            #pragma unroll
            for (int rr = 0; rr < 8; ++rr)
                pred[(r0 + mbase + rr) * 24 + col] = acc[rr] + bb;
        }
    }
}

// ---------------------------------------------------------------------------
extern "C" void kernel_launch(void* const* d_in, const int* in_sizes, int n_in,
                              void* d_out, int out_size, void* d_ws, size_t ws_size,
                              hipStream_t stream)
{
    const float* inputs      = (const float*)d_in[0];
    const int*   idxs        = (const int*)  d_in[2];
    const float* level_coef  = (const float*)d_in[3];
    const float* season_coef = (const float*)d_in[4];
    const float* seas_param  = (const float*)d_in[5];
    const float* Wih0        = (const float*)d_in[6];
    const float* WihR        = (const float*)d_in[7];   // (3,1024,256)
    const float* WhhA        = (const float*)d_in[8];   // (4,1024,256)
    const float* bvec        = (const float*)d_in[9];   // (4,1024)
    const float* nlW         = (const float*)d_in[10];
    const float* nlb         = (const float*)d_in[11];
    const float* scoreW      = (const float*)d_in[12];
    const float* scoreb      = (const float*)d_in[13];
    (void)in_sizes; (void)n_in; (void)out_size; (void)ws_size;

    char* ws = (char*)d_ws;
    size_t off = 0;
    auto alloc = [&](size_t bytes) -> char* {
        char* p = ws + off;
        off = (off + bytes + 255) & ~(size_t)255;
        return p;
    };
    float*  levels = (float*) alloc((size_t)256 * 720 * 4);
    float*  seas   = (float*) alloc((size_t)256 * 744 * 4);
    __bf16* xbuf   = (__bf16*)alloc((size_t)TLEN * 256 * 24 * 2);
    __bf16* h0     = (__bf16*)alloc((size_t)TLEN * 256 * 256 * 2);
    __bf16* h1     = (__bf16*)alloc((size_t)TLEN * 256 * 256 * 2);  // residual r
    __bf16* h2     = (__bf16*)alloc((size_t)TLEN * 256 * 256 * 2);
    __bf16* w0     = (__bf16*)alloc((size_t)64 * 9  * 512 * 2);
    __bf16* w1     = (__bf16*)alloc((size_t)64 * 16 * 512 * 2);
    __bf16* w2     = (__bf16*)alloc((size_t)64 * 16 * 512 * 2);
    __bf16* w3     = (__bf16*)alloc((size_t)64 * 16 * 512 * 2);
    __bf16* wnl    = (__bf16*)alloc((size_t)16 * 8  * 512 * 2);
    __bf16* wsc    = (__bf16*)alloc((size_t)2  * 8  * 512 * 2);

    float* pred   = (float*)d_out;
    float* actual = pred + (size_t)TLEN * 256 * 24;

    hw_kernel<<<1, 256, 0, stream>>>(inputs, idxs, level_coef, season_coef,
                                     seas_param, levels, seas);
    window_kernel<<<673, 256, 0, stream>>>(inputs, levels, seas, xbuf, actual);

    // pack weights into WMMA B-fragment layout (bf16)
    pack_kernel<<< 72, 256, 0, stream>>>(Wih0,            24, WhhA + 0*262144,  32,  9, 64, 1024, w0);
    pack_kernel<<<128, 256, 0, stream>>>(WihR + 0*262144, 256, WhhA + 1*262144, 256, 16, 64, 1024, w1);
    pack_kernel<<<128, 256, 0, stream>>>(WihR + 1*262144, 256, WhhA + 2*262144, 256, 16, 64, 1024, w2);
    pack_kernel<<<128, 256, 0, stream>>>(WihR + 2*262144, 256, WhhA + 3*262144, 256, 16, 64, 1024, w3);
    pack_kernel<<< 16, 256, 0, stream>>>(nlW,            256, nullptr,          256,  8, 16,  256, wnl);
    pack_kernel<<<  4, 256, 0, stream>>>(scoreW,         256, nullptr,          256,  8,  2,   24, wsc);

    // dilated LSTM stack (grid = dil*16 workgroups of 16 rows each)
    lstm_kernel<32> <<< 16, 256, 0, stream>>>(xbuf, w0, bvec + 0,    h0, 1);
    lstm_kernel<256><<< 32, 256, 0, stream>>>(h0,   w1, bvec + 1024, h1, 2);
    lstm_kernel<256><<< 64, 256, 0, stream>>>(h1,   w2, bvec + 2048, h2, 4);
    lstm_kernel<256><<<128, 256, 0, stream>>>(h2,   w3, bvec + 3072, h0, 8);

    // hf = tanh((h3 + r) @ nlW^T + nlb)  (reuse h2 buffer), then score head
    fuse_tanh_kernel<<<10768, 256, 0, stream>>>(h0, h1, wnl, nlb, h2);
    score_kernel<<<1346, 256, 0, stream>>>(h2, wsc, scoreb, pred);
}
